// Attention_17652315587390
// MI455X (gfx1250) — matrix operand order repaired and checked by reading, not compiled, not run on hardware
//
#include <hip/hip_runtime.h>
#include <hip/hip_bf16.h>
#include <stdint.h>

typedef __attribute__((ext_vector_type(16))) __bf16 v16bf;
typedef __attribute__((ext_vector_type(2)))  __bf16 bf16x2;
typedef __attribute__((ext_vector_type(8)))  float  v8f;

#define D_MODEL   1024
#define NUM_HEADS 16
#define D_KH      64
#define BATCH     2
#define SEQ       2048
#define M_TOTAL   (BATCH * SEQ)                                // 4096
#define TENS_ELEMS ((size_t)M_TOTAL * D_MODEL)                 // 4,194,304
#define HEAD_ELEMS ((size_t)BATCH * NUM_HEADS * SEQ * D_KH)    // 4,194,304

__device__ __forceinline__ unsigned short f2bf_bits(float f) {
    unsigned u = __builtin_bit_cast(unsigned, f);
    u += 0x7fffu + ((u >> 16) & 1u);            // round-to-nearest-even
    return (unsigned short)(u >> 16);
}

// gfx1250 async global->LDS copy (ASYNCcnt path); 16B per lane (+offset:16 pair).
__device__ __forceinline__ void async_b128(unsigned lds_addr, const void* gaddr) {
    asm volatile("global_load_async_to_lds_b128 %0, %1, off"
                 :: "v"(lds_addr), "v"(gaddr) : "memory");
}
__device__ __forceinline__ void async_b128_o16(unsigned lds_addr, const void* gaddr) {
    asm volatile("global_load_async_to_lds_b128 %0, %1, off offset:16"
                 :: "v"(lds_addr), "v"(gaddr) : "memory");
}
__device__ __forceinline__ void wait_async4() {
    asm volatile("s_wait_asynccnt 0x4" ::: "memory");
}
__device__ __forceinline__ void wait_async0() {
    asm volatile("s_wait_asynccnt 0x0" ::: "memory");
}

// ---------------------------------------------------------------------------
// Prologue 1: fp32 -> bf16 elementwise (q,k,v), 8 elems/thread.
// ---------------------------------------------------------------------------
__global__ __launch_bounds__(256)
void cvt_f32_bf16(const float* __restrict__ A0, const float* __restrict__ A1,
                  const float* __restrict__ A2, unsigned short* __restrict__ dst)
{
    const int z = blockIdx.z;
    const float* src = (z == 0) ? A0 : ((z == 1) ? A1 : A2);
    size_t base = ((size_t)blockIdx.x * 256 + threadIdx.x) * 8;
    float4 f0 = *(const float4*)(src + base);
    float4 f1 = *(const float4*)(src + base + 4);
    uint4 o;
    o.x = (unsigned)f2bf_bits(f0.x) | ((unsigned)f2bf_bits(f0.y) << 16);
    o.y = (unsigned)f2bf_bits(f0.z) | ((unsigned)f2bf_bits(f0.w) << 16);
    o.z = (unsigned)f2bf_bits(f1.x) | ((unsigned)f2bf_bits(f1.y) << 16);
    o.w = (unsigned)f2bf_bits(f1.z) | ((unsigned)f2bf_bits(f1.w) << 16);
    *(uint4*)(dst + z * TENS_ELEMS + base) = o;
}

// ---------------------------------------------------------------------------
// Prologue 2: W[k][n] fp32 -> Wt[n][k] bf16 (32x32 LDS tiles). z: Wq / Wo.
// ---------------------------------------------------------------------------
__global__ __launch_bounds__(256)
void transpose_w_bf16(const float* __restrict__ W0, const float* __restrict__ W1,
                      unsigned short* __restrict__ T0, unsigned short* __restrict__ T1)
{
    __shared__ unsigned short t[32][33];
    const float* W = blockIdx.z ? W1 : W0;
    unsigned short* T = blockIdx.z ? T1 : T0;
    const int tx = threadIdx.x & 31, ty = threadIdx.x >> 5;    // 32 x 8
    const int k0 = blockIdx.x * 32, n0 = blockIdx.y * 32;
#pragma unroll
    for (int i = 0; i < 32; i += 8)
        t[ty + i][tx] = f2bf_bits(W[(size_t)(k0 + ty + i) * D_MODEL + n0 + tx]);
    __syncthreads();
#pragma unroll
    for (int i = 0; i < 32; i += 8)
        T[(size_t)(n0 + ty + i) * D_MODEL + k0 + tx] = t[tx][ty + i];
}

// ---------------------------------------------------------------------------
// GEMM: C[4096,1024] = Abf[4096,1024](bf16) * Wt[n][k](bf16) + bias
// Async double-buffered LDS staging (peeled pipeline), WMMA f32_16x16x32_bf16.
// mode==1: bf16 head layout; z==0 -> Q (pre-scaled by 1/sqrt(dk)), z==1 -> K,
//          both [b][h][s][dk]; z==2 -> V transposed [b][h][dk][s].
// mode==0: fp32 row-major + bias.
// Block tile 128x128, 8 waves each 64x32 (4x2 frags).
// ---------------------------------------------------------------------------
__global__ __launch_bounds__(256)
void gemm_wmma_bf16(const unsigned short* __restrict__ A0,
                    const unsigned short* __restrict__ A1,
                    const unsigned short* __restrict__ A2,
                    const unsigned short* __restrict__ Wt,
                    const float* __restrict__ bias, float* __restrict__ outF,
                    unsigned short* __restrict__ outH, int mode)
{
    __shared__ __align__(16) unsigned short As[2][128][40]; // [m][k]
    __shared__ __align__(16) unsigned short Bs[2][128][40]; // [n][k]

    const int tid  = threadIdx.x;
    const int wave = tid >> 5, lane = tid & 31;
    const int half = lane >> 4, ln = lane & 15;
    const int m0 = blockIdx.y * 128, n0 = blockIdx.x * 128;
    const int z  = blockIdx.z;
    const unsigned short* Ab = (z == 0) ? A0 : ((z == 1) ? A1 : A2);

    const int MW = (wave & 1) * 64;
    const int NW = (wave >> 1) * 32;
    const int srow = tid >> 1, scol = (tid & 1) * 16;   // staging: 2 thr/row

    v8f vzero = {};
    v8f acc[4][2];
#pragma unroll
    for (int i = 0; i < 4; ++i)
#pragma unroll
        for (int j = 0; j < 2; ++j) acc[i][j] = vzero;

    auto issue_tiles = [&](int buf, int k0) {
        const unsigned short* ga = Ab + (size_t)(m0 + srow) * D_MODEL + k0 + scol;
        unsigned la = (unsigned)(uintptr_t)&As[buf][srow][scol];
        async_b128(la, ga);
        async_b128_o16(la, ga);
        const unsigned short* gb = Wt + (size_t)(n0 + srow) * D_MODEL + k0 + scol;
        unsigned lb = (unsigned)(uintptr_t)&Bs[buf][srow][scol];
        async_b128(lb, gb);
        async_b128_o16(lb, gb);
    };

    auto compute_tile = [&](int buf) {
        v16bf a[4], b[2];
#pragma unroll
        for (int i = 0; i < 4; ++i)
#pragma unroll
            for (int j = 0; j < 8; ++j) {
                int kk = 2 * j + (j >= 4 ? 8 : 0) + 8 * half;     // A K-pattern
                bf16x2 p = __builtin_bit_cast(bf16x2,
                            *(const unsigned*)&As[buf][MW + 16 * i + ln][kk]);
                a[i][2 * j] = p.x; a[i][2 * j + 1] = p.y;
            }
#pragma unroll
        for (int jn = 0; jn < 2; ++jn)
#pragma unroll
            for (int j = 0; j < 8; ++j) {
                int kk = 2 * j + (j >= 4 ? 8 : 0) + 8 * half;     // B K-pattern
                bf16x2 p = __builtin_bit_cast(bf16x2,
                            *(const unsigned*)&Bs[buf][NW + 16 * jn + ln][kk]);
                b[jn][2 * j] = p.x; b[jn][2 * j + 1] = p.y;
            }
#pragma unroll
        for (int i = 0; i < 4; ++i)
#pragma unroll
            for (int jn = 0; jn < 2; ++jn)
                acc[i][jn] = __builtin_amdgcn_wmma_f32_16x16x32_bf16(
                    false, a[i], false, b[jn], (short)0, acc[i][jn], false, false);
    };

    // software pipeline, last iteration peeled (branch-free steady state)
    issue_tiles(0, 0);
    int buf = 0;
    for (int k0 = 0; k0 < D_MODEL - 32; k0 += 32, buf ^= 1) {
        issue_tiles(buf ^ 1, k0 + 32);
        wait_async4();
        __syncthreads();
        compute_tile(buf);
        __syncthreads();
    }
    wait_async0();
    __syncthreads();
    compute_tile(buf);

#pragma unroll
    for (int i = 0; i < 4; ++i)
#pragma unroll
        for (int jn = 0; jn < 2; ++jn)
#pragma unroll
            for (int r = 0; r < 8; ++r) {
                int gm = m0 + MW + 16 * i + r + 8 * half;
                int gn = n0 + NW + 16 * jn + ln;
                float v = acc[i][jn][r] + bias[gn];
                if (mode == 0) {
                    outF[(size_t)gm * D_MODEL + gn] = v;
                } else {
                    if (z == 0) v *= 0.125f;   // fold 1/sqrt(dk) into Q
                    int b_ = gm >> 11, s_ = gm & 2047;
                    int h_ = gn >> 6,  dk = gn & 63;
                    size_t off;
                    if (z == 2)  // V stored [b][h][dk][s] for async tile loads
                        off = ((size_t)(b_ * NUM_HEADS + h_) * D_KH + dk) * SEQ + s_;
                    else
                        off = ((size_t)(b_ * NUM_HEADS + h_) * SEQ + s_) * D_KH + dk;
                    outH[off + (size_t)z * HEAD_ELEMS] = f2bf_bits(v);
                }
            }
}

// ---------------------------------------------------------------------------
// Flash attention: block = 128 query rows of one (b,h); wave owns 16 rows.
// K/V tiles async double-buffered in LDS (peeled pipeline); online softmax in
// VGPRs; P re-laid out via per-wave LDS scratch; ctx written bf16.
// ---------------------------------------------------------------------------
__global__ __launch_bounds__(256)
void flash_attn_wmma(const unsigned short* __restrict__ Qh,
                     const unsigned short* __restrict__ Kh,
                     const unsigned short* __restrict__ Vth,
                     unsigned short* __restrict__ ctx)
{
    __shared__ __align__(16) unsigned short Ks[2][64][72];   // [t][dk]
    __shared__ __align__(16) unsigned short Vts[2][64][72];  // [dk][t]
    __shared__ __align__(16) unsigned short Ps[8][16][72];   // per-wave P

    const int tid  = threadIdx.x;
    const int wave = tid >> 5, lane = tid & 31;
    const int half = lane >> 4, ln = lane & 15;
    const int bh = blockIdx.y;
    const int q0 = blockIdx.x * 128 + wave * 16;

    const unsigned short* Qp  = Qh  + (size_t)bh * SEQ * D_KH;
    const unsigned short* Kp  = Kh  + (size_t)bh * SEQ * D_KH;
    const unsigned short* Vtp = Vth + (size_t)bh * D_KH * SEQ;

    // resident Q A-fragments (two K=32 steps over dk); Q is pre-scaled
    v16bf qa[2];
#pragma unroll
    for (int ts = 0; ts < 2; ++ts)
#pragma unroll
        for (int j = 0; j < 8; ++j) {
            int kk = 32 * ts + 2 * j + (j >= 4 ? 8 : 0) + 8 * half;
            bf16x2 p = __builtin_bit_cast(bf16x2,
                        *(const unsigned*)(Qp + (size_t)(q0 + ln) * D_KH + kk));
            qa[ts][2 * j] = p.x; qa[ts][2 * j + 1] = p.y;
        }

    v8f vzero = {};
    v8f o[4];
#pragma unroll
    for (int jd = 0; jd < 4; ++jd) o[jd] = vzero;
    float mrow[8], lrow[8];
#pragma unroll
    for (int r = 0; r < 8; ++r) { mrow[r] = -1e30f; lrow[r] = 0.0f; }

    const int krow = tid >> 2;          // 0..63
    const int kseg = (tid & 3) * 16;    // 0,16,32,48

    auto issue_tiles = [&](int buf, int kt) {
        const unsigned short* gk = Kp + (size_t)(kt + krow) * D_KH + kseg;
        unsigned lk = (unsigned)(uintptr_t)&Ks[buf][krow][kseg];
        async_b128(lk, gk);
        async_b128_o16(lk, gk);
        const unsigned short* gv = Vtp + (size_t)krow * SEQ + kt + kseg;
        unsigned lv = (unsigned)(uintptr_t)&Vts[buf][krow][kseg];
        async_b128(lv, gv);
        async_b128_o16(lv, gv);
    };

    auto process_tile = [&](int buf) {
        // scores = Q * K^T  (Q pre-scaled by 1/sqrt(dk))
        v8f sc[4];
#pragma unroll
        for (int jn = 0; jn < 4; ++jn) sc[jn] = vzero;
#pragma unroll
        for (int jn = 0; jn < 4; ++jn)
#pragma unroll
            for (int ts = 0; ts < 2; ++ts) {
                v16bf bfrag;
#pragma unroll
                for (int j = 0; j < 8; ++j) {
                    int kk = 32 * ts + 2 * j + (j >= 4 ? 8 : 0) + 8 * half;
                    bf16x2 p = __builtin_bit_cast(bf16x2,
                                *(const unsigned*)&Ks[buf][16 * jn + ln][kk]);
                    bfrag[2 * j] = p.x; bfrag[2 * j + 1] = p.y;
                }
                sc[jn] = __builtin_amdgcn_wmma_f32_16x16x32_bf16(
                    false, qa[ts], false, bfrag, (short)0, sc[jn], false, false);
            }

        // online softmax
        float alpha[8];
#pragma unroll
        for (int r = 0; r < 8; ++r) {
            float x = fmaxf(fmaxf(sc[0][r], sc[1][r]), fmaxf(sc[2][r], sc[3][r]));
#pragma unroll
            for (int msk = 1; msk < 16; msk <<= 1)
                x = fmaxf(x, __shfl_xor(x, msk, 16));
            float mnew = fmaxf(mrow[r], x);
            alpha[r] = __expf(mrow[r] - mnew);
            mrow[r] = mnew;
        }
        float rs[8];
#pragma unroll
        for (int r = 0; r < 8; ++r) rs[r] = 0.0f;
#pragma unroll
        for (int jn = 0; jn < 4; ++jn)
#pragma unroll
            for (int r = 0; r < 8; ++r) {
                float p = __expf(sc[jn][r] - mrow[r]);
                sc[jn][r] = p;
                rs[r] += p;
            }
#pragma unroll
        for (int r = 0; r < 8; ++r) {
#pragma unroll
            for (int msk = 1; msk < 16; msk <<= 1)
                rs[r] += __shfl_xor(rs[r], msk, 16);
            lrow[r] = lrow[r] * alpha[r] + rs[r];
#pragma unroll
            for (int jd = 0; jd < 4; ++jd) o[jd][r] *= alpha[r];
        }

        // P: C layout -> per-wave LDS -> A layout
#pragma unroll
        for (int jn = 0; jn < 4; ++jn)
#pragma unroll
            for (int r = 0; r < 8; ++r)
                Ps[wave][r + 8 * half][16 * jn + ln] = f2bf_bits(sc[jn][r]);

        v16bf pa[2];
#pragma unroll
        for (int ts = 0; ts < 2; ++ts)
#pragma unroll
            for (int j = 0; j < 8; ++j) {
                int kk = 32 * ts + 2 * j + (j >= 4 ? 8 : 0) + 8 * half;
                bf16x2 p = __builtin_bit_cast(bf16x2,
                            *(const unsigned*)&Ps[wave][ln][kk]);
                pa[ts][2 * j] = p.x; pa[ts][2 * j + 1] = p.y;
            }

        // O += P * V
#pragma unroll
        for (int jd = 0; jd < 4; ++jd)
#pragma unroll
            for (int ts = 0; ts < 2; ++ts) {
                v16bf bfrag;
#pragma unroll
                for (int j = 0; j < 8; ++j) {
                    int kk = 32 * ts + 2 * j + (j >= 4 ? 8 : 0) + 8 * half;
                    bf16x2 p = __builtin_bit_cast(bf16x2,
                                *(const unsigned*)&Vts[buf][16 * jd + ln][kk]);
                    bfrag[2 * j] = p.x; bfrag[2 * j + 1] = p.y;
                }
                o[jd] = __builtin_amdgcn_wmma_f32_16x16x32_bf16(
                    false, pa[ts], false, bfrag, (short)0, o[jd], false, false);
            }
    };

    // software pipeline over key tiles, last iteration peeled
    issue_tiles(0, 0);
    int buf = 0;
    for (int kt = 0; kt < SEQ - 64; kt += 64, buf ^= 1) {
        issue_tiles(buf ^ 1, kt + 64);
        wait_async4();
        __syncthreads();
        process_tile(buf);
        __syncthreads();
    }
    wait_async0();
    __syncthreads();
    process_tile(buf);

    // epilogue: O / l -> ctx bf16 [b][s][h*64+dk]
    const int b_ = bh >> 4, h_ = bh & 15;
#pragma unroll
    for (int jd = 0; jd < 4; ++jd)
#pragma unroll
        for (int r = 0; r < 8; ++r) {
            int row = q0 + r + 8 * half;
            float v = o[jd][r] / lrow[r];
            ctx[(size_t)(b_ * SEQ + row) * D_MODEL + h_ * D_KH + 16 * jd + ln]
                = f2bf_bits(v);
        }
}

// ---------------------------------------------------------------------------
extern "C" void kernel_launch(void* const* d_in, const int* in_sizes, int n_in,
                              void* d_out, int out_size, void* d_ws, size_t ws_size,
                              hipStream_t stream) {
    (void)in_sizes; (void)n_in; (void)out_size; (void)ws_size;
    const float* q  = (const float*)d_in[0];
    const float* k  = (const float*)d_in[1];
    const float* v  = (const float*)d_in[2];
    const float* Wq = (const float*)d_in[3];
    const float* bq = (const float*)d_in[4];
    const float* Wo = (const float*)d_in[5];
    const float* bo = (const float*)d_in[6];
    float* out = (float*)d_out;

    // ws layout (bf16 elems unless noted):
    //  [0)        Aq (8MB)  -- reused later as ctx (bf16, 8MB)
    //  [8MB)      Ak, [16MB) Av
    //  [24MB)     Wqt (2MB), [26MB) Wot (2MB)
    //  [28MB)     Qh, [36MB) Kh, [44MB) Vth   (8MB each) -> 52MB total
    unsigned short* Aq  = (unsigned short*)d_ws;
    unsigned short* Ak  = Aq + TENS_ELEMS;
    unsigned short* Av  = Ak + TENS_ELEMS;
    unsigned short* Wqt = Av + TENS_ELEMS;
    unsigned short* Wot = Wqt + (size_t)D_MODEL * D_MODEL;
    unsigned short* Qh  = Wot + (size_t)D_MODEL * D_MODEL;
    unsigned short* ctx = Aq;   // Aq dead after QKV projection

    dim3 blk(256);
    // 0) fp32 -> bf16 activations; weight transposes
    cvt_f32_bf16<<<dim3((unsigned)(TENS_ELEMS / (256 * 8)), 1, 3), blk, 0, stream>>>(
        q, k, v, Aq);
    transpose_w_bf16<<<dim3(32, 32, 2), blk, 0, stream>>>(Wq, Wo, Wqt, Wot);
    // 1) QKV projection (shared Wq/bq per reference); Q pre-scaled, V transposed
    gemm_wmma_bf16<<<dim3(D_MODEL / 128, M_TOTAL / 128, 3), blk, 0, stream>>>(
        Aq, Ak, Av, Wqt, bq, nullptr, Qh, 1);
    // 2) flash attention
    flash_attn_wmma<<<dim3(SEQ / 128, BATCH * NUM_HEADS), blk, 0, stream>>>(
        Qh, Qh + HEAD_ELEMS, Qh + 2 * HEAD_ELEMS, ctx);
    // 3) output projection (fp32 out + bias)
    gemm_wmma_bf16<<<dim3(D_MODEL / 128, M_TOTAL / 128, 1), blk, 0, stream>>>(
        ctx, ctx, ctx, Wot, bo, out, nullptr, 0);
}
